// TransformerLM_34806414967115
// MI455X (gfx1250) — matrix operand order repaired and checked by reading, not compiled, not run on hardware
//
#include <hip/hip_runtime.h>
#include <cstdint>
#include <cstddef>

// ---------------------------------------------------------------------------
// CDNA5 (gfx1250) LM-head + cross-entropy.
//   logits = h[M,D] * W[V,D]^T + b[V]  ;  loss = mean(nll(log_softmax(logits), y))
// bf16 WMMA (V_WMMA_F32_16X16X32_BF16), f32 accumulate.
// ---------------------------------------------------------------------------

typedef __attribute__((ext_vector_type(16))) __bf16 v16bf;
typedef __attribute__((ext_vector_type(8)))  __bf16 v8bf;
typedef __attribute__((ext_vector_type(8)))  float  v8f;

#define TILE_M 128
#define TILE_N 128
#define TILE_K 32
#define LDS_STRIDE 40   // 32 + 8 bf16 pad -> 80B row stride, conflict-free ds_load_b128

__device__ __forceinline__ unsigned short f32_to_bf16(float f) {
  unsigned int u = __float_as_uint(f);
  u += 0x7FFFu + ((u >> 16) & 1u);     // round-to-nearest-even
  return (unsigned short)(u >> 16);
}

// ---------------------------------------------------------------------------
// Kernel 1: f32 -> bf16 conversion (vectorized), W becomes L2-resident (103MB)
// ---------------------------------------------------------------------------
__global__ __launch_bounds__(256)
void convert_f32_to_bf16(const float* __restrict__ src,
                         unsigned short* __restrict__ dst, int n4) {
  int i = blockIdx.x * blockDim.x + threadIdx.x;
  if (i < n4) {
    float4 f = reinterpret_cast<const float4*>(src)[i];
    ushort4 u;
    u.x = f32_to_bf16(f.x);
    u.y = f32_to_bf16(f.y);
    u.z = f32_to_bf16(f.z);
    u.w = f32_to_bf16(f.w);
    reinterpret_cast<ushort4*>(dst)[i] = u;
  }
}

// ---------------------------------------------------------------------------
// Kernel 2: WMMA GEMM  out[M,V] = A[M,D](bf16) * W[V,D](bf16)^T + bias
//   block = 256 threads = 8 waves (wave32), block tile 128x128
//   wave grid 2x4, wave tile 64x32 -> 4x2 accumulators of 16x16
// ---------------------------------------------------------------------------
__global__ __launch_bounds__(256)
void lmhead_gemm_bf16(const unsigned short* __restrict__ Ab,  // [M,D] bf16 bits
                      const unsigned short* __restrict__ Wb,  // [V,D] bf16 bits
                      const float* __restrict__ bias,         // [V]
                      float* __restrict__ out,                // [M,V]
                      int M, int D, int V) {
  __shared__ unsigned short As[TILE_M * LDS_STRIDE];

  const int tid  = threadIdx.x;
  const int wave = tid >> 5;
  const int lane = tid & 31;
  const int nl   = lane & 15;   // sub-tile row/col index within fragment
  const int hl   = lane >> 4;   // half-of-wave select
  const int wm   = wave >> 2;   // 0..1  (wave row)
  const int wn   = wave & 3;    // 0..3  (wave col)

  const int row0 = blockIdx.y * TILE_M;
  const int col0 = blockIdx.x * TILE_N;

  const v8f vzero = {0.f, 0.f, 0.f, 0.f, 0.f, 0.f, 0.f, 0.f};
  v8f acc[4][2];
#pragma unroll
  for (int mt = 0; mt < 4; ++mt)
#pragma unroll
    for (int nt = 0; nt < 2; ++nt) acc[mt][nt] = vzero;

  // Precompute clamped B columns (edge N-tile: clamp loads, guard stores,
  // keeps EXEC all-1s as WMMA requires).
  int colc[2];
#pragma unroll
  for (int nt = 0; nt < 2; ++nt) {
    int col = col0 + wn * 32 + nt * 16 + nl;
    colc[nt] = (col < V) ? col : (V - 1);
  }

  for (int k0 = 0; k0 < D; k0 += TILE_K) {
    // ---- stage A tile (128 x 32 bf16) into LDS: 512 x 16B chunks ----
#pragma unroll
    for (int it = 0; it < 2; ++it) {
      int chunk = tid + it * 256;          // 0..511
      int r  = chunk >> 2;                 // 0..127
      int c8 = chunk & 3;                  // 16B chunk within row
      int gr = row0 + r; if (gr >= M) gr = M - 1;
      v8bf v = *reinterpret_cast<const v8bf*>(Ab + (size_t)gr * D + k0 + c8 * 8);
      *reinterpret_cast<v8bf*>(&As[r * LDS_STRIDE + c8 * 8]) = v;
    }
    __syncthreads();

    // ---- B fragments straight from global (L2-resident bf16 W) ----
    // B layout used here (same as A): lane h*16+n holds K = 16h..16h+15 of column n,
    // which is 32 contiguous bytes of row-major W[col].
    v16bf bfrag[2];
#pragma unroll
    for (int nt = 0; nt < 2; ++nt) {
      const v8bf* pB =
          reinterpret_cast<const v8bf*>(Wb + (size_t)colc[nt] * D + k0 + hl * 16);
      v8bf blo = pB[0];
      v8bf bhi = pB[1];
      if (k0 + TILE_K < D)
        __builtin_prefetch(Wb + (size_t)colc[nt] * D + k0 + TILE_K, 0, 0);
      bfrag[nt] = __builtin_shufflevector(blo, bhi,
          0, 1, 2, 3, 4, 5, 6, 7, 8, 9, 10, 11, 12, 13, 14, 15);
    }

    // ---- A fragments from LDS + 8 WMMAs ----
    // A layout (matches B): lane h*16+m holds K = 16h..16h+15
    // of row m -> two ds_load_b128 per fragment.
#pragma unroll
    for (int mt = 0; mt < 4; ++mt) {
      int mrow = wm * 64 + mt * 16 + nl;
      const v8bf* pA = reinterpret_cast<const v8bf*>(&As[mrow * LDS_STRIDE]);
      v8bf alo = pA[2 * hl];        // K = 16h .. 16h+7
      v8bf ahi = pA[2 * hl + 1];    // K = 16h+8 .. 16h+15
      v16bf afrag = __builtin_shufflevector(alo, ahi,
          0, 1, 2, 3, 4, 5, 6, 7, 8, 9, 10, 11, 12, 13, 14, 15);
#pragma unroll
      for (int nt = 0; nt < 2; ++nt) {
        acc[mt][nt] = __builtin_amdgcn_wmma_f32_16x16x32_bf16(
            /*neg_a=*/false, afrag, /*neg_b=*/false, bfrag[nt],
            /*c_mod=*/(short)0, acc[mt][nt],
            /*reuse_a=*/false, /*reuse_b=*/false);
      }
    }
    __syncthreads();
  }

  // ---- epilogue: fused bias add + store ----
  // C/D layout: VGPR r of lane l holds element (m = r + 8*(l>>4), n = l&15).
#pragma unroll
  for (int nt = 0; nt < 2; ++nt) {
    int col = col0 + wn * 32 + nt * 16 + nl;
    if (col >= V) continue;
    float bv = bias[col];
#pragma unroll
    for (int mt = 0; mt < 4; ++mt) {
      int rbase = row0 + wm * 64 + mt * 16 + hl * 8;
#pragma unroll
      for (int r = 0; r < 8; ++r) {
        int grow = rbase + r;
        if (grow < M) out[(size_t)grow * V + col] = acc[mt][nt][r] + bv;
      }
    }
  }
}

// ---------------------------------------------------------------------------
// Kernel 3: per-row online log-sum-exp + NLL.  One workgroup per row.
// Single pass over the 823MB logits (online max/sum), aligned float4 body.
// ---------------------------------------------------------------------------
__global__ __launch_bounds__(256)
void row_nll(const float* __restrict__ logits, const int* __restrict__ y,
             float* __restrict__ nll, int V) {
  const int row = blockIdx.x;
  const int tid = threadIdx.x;
  const float* rowp = logits + (size_t)row * V;

  float m = -__builtin_inff();
  float s = 0.f;

  auto upd = [&](float x) {
    if (x > m) { s = s * __expf(m - x) + 1.0f; m = x; }
    else       { s += __expf(x - m); }
  };

  // scalar head until 16B alignment of (row*V + head)
  int head = (4 - (int)(((size_t)row * (size_t)V) & 3)) & 3;
  if (head > V) head = V;
  for (int j = tid; j < head; j += 256) upd(rowp[j]);

  const int nv = (V - head) >> 2;
  const float4* vp = reinterpret_cast<const float4*>(rowp + head);
  for (int i = tid; i < nv; i += 256) {
    float4 f = vp[i];
    upd(f.x); upd(f.y); upd(f.z); upd(f.w);
  }
  for (int j = head + nv * 4 + tid; j < V; j += 256) upd(rowp[j]);

  // tree-combine (m, s) pairs
  __shared__ float sm[256];
  __shared__ float ss[256];
  sm[tid] = m; ss[tid] = s;
  __syncthreads();
  for (int off = 128; off > 0; off >>= 1) {
    if (tid < off) {
      float m1 = sm[tid],        s1 = ss[tid];
      float m2 = sm[tid + off],  s2 = ss[tid + off];
      float mn = fmaxf(m1, m2);
      sm[tid] = mn;
      ss[tid] = s1 * __expf(m1 - mn) + s2 * __expf(m2 - mn);
    }
    __syncthreads();
  }

  if (tid == 0) {
    int t = y[row];
    nll[row] = sm[0] + __logf(ss[0]) - rowp[t];
  }
}

// ---------------------------------------------------------------------------
// Kernel 4: mean(nll) -> loss
// ---------------------------------------------------------------------------
__global__ __launch_bounds__(256)
void mean_reduce(const float* __restrict__ nll, float* __restrict__ loss, int M) {
  const int tid = threadIdx.x;
  float sum = 0.f;
  for (int i = tid; i < M; i += 256) sum += nll[i];
  __shared__ float sh[256];
  sh[tid] = sum;
  __syncthreads();
  for (int off = 128; off > 0; off >>= 1) {
    if (tid < off) sh[tid] += sh[tid + off];
    __syncthreads();
  }
  if (tid == 0) loss[0] = sh[0] / (float)M;
}

// ---------------------------------------------------------------------------
// Launch
// ---------------------------------------------------------------------------
extern "C" void kernel_launch(void* const* d_in, const int* in_sizes, int n_in,
                              void* d_out, int out_size, void* d_ws, size_t ws_size,
                              hipStream_t stream) {
  const float* h = (const float*)d_in[0];   // [M, D]
  const float* W = (const float*)d_in[1];   // [V, D]
  const float* b = (const float*)d_in[2];   // [V]
  const int*   y = (const int*)d_in[3];     // [M]

  const int V = in_sizes[2];
  const int D = in_sizes[1] / V;
  const int M = in_sizes[0] / D;

  float* loss   = (float*)d_out;
  float* logits = (float*)d_out + 1;

  // workspace layout: h_bf16 | W_bf16 | nll
  char* ws = (char*)d_ws;
  unsigned short* hb = (unsigned short*)ws;
  unsigned short* Wb = hb + (size_t)M * D;
  size_t off = ((size_t)M * D + (size_t)V * D) * sizeof(unsigned short);
  off = (off + 15) & ~(size_t)15;
  float* nll = (float*)(ws + off);

  // 1) convert to bf16 (W becomes 103MB -> L2-resident)
  int n4h = (M * D) / 4;
  convert_f32_to_bf16<<<(n4h + 255) / 256, 256, 0, stream>>>(h, hb, n4h);
  int n4w = (int)(((size_t)V * D) / 4);
  convert_f32_to_bf16<<<(n4w + 255) / 256, 256, 0, stream>>>(W, Wb, n4w);

  // 2) WMMA GEMM + bias
  dim3 grid((V + TILE_N - 1) / TILE_N, (M + TILE_M - 1) / TILE_M);
  lmhead_gemm_bf16<<<grid, 256, 0, stream>>>(hb, Wb, b, logits, M, D, V);

  // 3) per-row log-softmax NLL (single pass)
  row_nll<<<M, 256, 0, stream>>>(logits, y, nll, V);

  // 4) mean -> loss
  mean_reduce<<<1, 256, 0, stream>>>(nll, loss, M);
}
